// RecurrentChannelAttention_50998441672740
// MI455X (gfx1250) — compile-verified, matrix operand order checked by reading, not run on hardware
//
#include <hip/hip_runtime.h>
#include <hip/hip_bf16.h>
#include <stdint.h>

// ---------------- problem constants (match reference) ----------------
#define C_DIM 512
#define T_DIM 8192
#define M_DIM 5
#define H_DIM 16
#define MOM   0.05f

#define NST   8                 // async ring stages (deeper = more bytes in flight)
#define TB    16                // output staging depth (timesteps)
#define KV_F  (C_DIM * M_DIM)   // 2560 floats per timestep for K or V
#define ASYNC_PER_STAGE 3       // async-load instructions issued per wave per stage

typedef float  v8f   __attribute__((ext_vector_type(8)));
typedef float  Vec4  __attribute__((ext_vector_type(4)));
typedef int    v4i   __attribute__((ext_vector_type(4)));
typedef __bf16 v16bf __attribute__((ext_vector_type(16)));

typedef __attribute__((address_space(1))) v4i* gptr_v4i;
typedef __attribute__((address_space(3))) v4i* lptr_v4i;

// ---------------- async copy primitive (global -> LDS) ----------------
#if __has_builtin(__builtin_amdgcn_global_load_async_to_lds_b128)
#define HAVE_ASYNC 1
__device__ __forceinline__ void cp16(void* lds, const void* gm) {
  __builtin_amdgcn_global_load_async_to_lds_b128(
      (gptr_v4i)(uintptr_t)gm,
      (lptr_v4i)(uint32_t)(uintptr_t)lds,
      0, 0);
}
#else
#define HAVE_ASYNC 0
__device__ __forceinline__ void cp16(void* lds, const void* gm) {
  *(Vec4*)lds = *(const Vec4*)gm;   // synchronous fallback
}
#endif

__device__ __forceinline__ void wait_async_stages() {
#if HAVE_ASYNC
#if __has_builtin(__builtin_amdgcn_s_wait_asynccnt)
  __builtin_amdgcn_s_wait_asynccnt((unsigned short)(ASYNC_PER_STAGE * (NST - 1)));
#else
  asm volatile("s_wait_asynccnt %0" :: "i"(ASYNC_PER_STAGE * (NST - 1)));
#endif
#endif
}

// Fill one ring stage. Each wave loads EXACTLY the region it later consumes
// (K,V floats [160w,160w+160), y floats [32w,32w+32)), so no barrier is needed
// between fill and consume: s_wait_asynccnt on the wave's own counter suffices.
__device__ __forceinline__ void stage_fill(int t, int w, int lane,
    const float* __restrict__ Kg, const float* __restrict__ Vg,
    const float* __restrict__ yT,
    float* ringK, float* ringV, float* ringY)
{
  const int slot = t & (NST - 1);
  const char* gk = (const char*)(Kg + (size_t)t * KV_F  + w * 160);
  const char* gv = (const char*)(Vg + (size_t)t * KV_F  + w * 160);
  const char* gy = (const char*)(yT + (size_t)t * C_DIM + w * 32);
  char* lk = (char*)(ringK + slot * KV_F  + w * 160);
  char* lv = (char*)(ringV + slot * KV_F  + w * 160);
  char* ly = (char*)(ringY + slot * C_DIM + w * 32);
  // per-wave region = 40 (K) + 40 (V) + 8 (y) float4s = 88; 3 passes of 32 lanes
#pragma unroll
  for (int jj = 0; jj < 3; ++jj) {
    int j = lane + 32 * jj;
    // branchless per-lane source/dest selection -> one async instr per pass
    const char* g = (j < 40) ? (gk + 16 * j)
                  : (j < 80) ? (gv + 16 * (j - 40))
                             : (gy + 16 * (j - 80));
    char*       l = (j < 40) ? (lk + 16 * j)
                  : (j < 80) ? (lv + 16 * (j - 40))
                             : (ly + 16 * (j - 80));
    if (j < 88) cp16(l, g);
  }
}

// ---------------- kernel 1: transpose y (1,C,T) -> yT (T,C) ----------------
__global__ void transpose_y_kernel(const float* __restrict__ y,
                                   float* __restrict__ yT)
{
  __shared__ float tile[32][33];
  const int tb = blockIdx.x * 32, cb = blockIdx.y * 32;
  const int tx = threadIdx.x, ty = threadIdx.y;
#pragma unroll
  for (int r = 0; r < 4; ++r)
    tile[ty + 8 * r][tx] = y[(size_t)(cb + ty + 8 * r) * T_DIM + tb + tx];
  __syncthreads();
#pragma unroll
  for (int r = 0; r < 4; ++r)
    yT[(size_t)(tb + ty + 8 * r) * C_DIM + cb + tx] = tile[tx][ty + 8 * r];
}

// ---------------- kernel 2: persistent sequential scan ----------------
__global__ __launch_bounds__(512, 1)
void ca_scan_kernel(const float* __restrict__ yT, const float* __restrict__ Kg,
                    const float* __restrict__ Vg, const float* __restrict__ W1,
                    const float* __restrict__ b1, const float* __restrict__ W2,
                    const float* __restrict__ b2, float* __restrict__ out)
{
  __shared__ float ringK[NST * KV_F];        // 80 KB
  __shared__ float ringV[NST * KV_F];        // 80 KB
  __shared__ float ringY[NST * C_DIM];       // 16 KB
  __shared__ float hpart[16 * H_DIM];
  __shared__ float rhbuf[H_DIM];
  __shared__ float wsum[16];
  __shared__ float outstage[C_DIM * (TB + 1)];  // 34 KB, padded vs bank conflicts

  const int tid  = threadIdx.x;
  const int w    = tid >> 5;       // wave 0..15, owns channels [32w,32w+32)
  const int lane = tid & 31;
  const int c    = tid;            // this thread's channel
  const int hi   = lane >> 4;
  const int jrow = lane & 15;

  // --- A-matrix: W1^T chunk (16 x 32) for this wave, bf16 WMMA A layout.
  // lanes 0-15 hold rows j; element e -> K = (e<8 ? e : e+8) + 8*hi.
  union { v16bf v; __bf16 e[16]; } Amat;
  const int c0 = w * 32;
#pragma unroll
  for (int e = 0; e < 16; ++e) {
    int K = (e < 8 ? e : e + 8) + 8 * hi;
    Amat.e[e] = (__bf16)W1[(c0 + K) * H_DIM + jrow];
  }
  float w2col[H_DIM];
#pragma unroll
  for (int j = 0; j < H_DIM; ++j) w2col[j] = W2[j * C_DIM + c];
  const float b2c = b2[c];
  const float b1j = (tid < H_DIM) ? b1[tid] : 0.0f;

  // prime the async ring
  for (int s = 0; s < NST; ++s)
    stage_fill(s, w, lane, Kg, Vg, yT, ringK, ringV, ringY);

  float yprev = 0.0f, pw = 1.0f;
  float ema[M_DIM];
#pragma unroll
  for (int m = 0; m < M_DIM; ++m) ema[m] = 0.0f;

  for (int t = 0; t < T_DIM; ++t) {
    const int slot = t & (NST - 1);
    wait_async_stages();                 // this wave's stage t is resident
    asm volatile("" ::: "memory");

    // ---- consume the whole ring slot into registers up front ----
    float sval = yprev + ringY[slot * C_DIM + c];
    const float* kp = &ringK[slot * KV_F + c * M_DIM];
    const float* vp = &ringV[slot * KV_F + c * M_DIM];
    float kf[M_DIM], vf[M_DIM];
#pragma unroll
    for (int m = 0; m < M_DIM; ++m) { kf[m] = kp[m]; vf[m] = vp[m]; }
    // ensure the DS reads above have landed, then refill this slot ASAP so the
    // async DMA overlaps the rest of the step (WMMA + 3 barriers).
    asm volatile("s_wait_dscnt 0x0" ::: "memory");
    const int tn = t + NST;
    if (tn < T_DIM)
      stage_fill(tn, w, lane, Kg, Vg, yT, ringK, ringV, ringY);

    // --- build B (32x16, every column = s chunk) in bf16 WMMA B layout
    // using only in-wave permutes (no LDS traffic, no barrier).
    float sx = __shfl_xor(sval, 1, 32);
    __bf16 blo = (__bf16)((lane & 1) ? sx : sval);
    __bf16 bhi = (__bf16)((lane & 1) ? sval : sx);
    unsigned pairv = (unsigned)__builtin_bit_cast(unsigned short, blo)
                   | ((unsigned)__builtin_bit_cast(unsigned short, bhi) << 16);
    union { v16bf v; unsigned u[8]; } Bmat;
#pragma unroll
    for (int p = 0; p < 8; ++p) {
      int idx = 4 * (2 * p + 8 * hi + (p >= 4 ? 8 : 0)); // src lane * 4 bytes
      Bmat.u[p] = (unsigned)__builtin_amdgcn_ds_bpermute(idx, (int)pairv);
    }

    // --- MLP layer 1 partial: P_w[j] = sum_{k<32} W1[c0+k, j] * s[c0+k]
    v8f acc = {};
    acc = __builtin_amdgcn_wmma_f32_16x16x32_bf16(false, Amat.v, false, Bmat.v,
                                                  (short)0, acc, false, false);
    // every column of D is identical: lane0 VGPR r = P[r], lane16 VGPR r = P[8+r]
    if (lane == 0 || lane == 16) {
      int base = w * H_DIM + hi * 8;
#pragma unroll
      for (int r = 0; r < 8; ++r) hpart[base + r] = acc[r];
    }
    __syncthreads();                                           // B1

    if (tid < H_DIM) {
      // 4 independent accumulator chains -> short dependence depth
      float s0 = 0.f, s1 = 0.f, s2 = 0.f, s3 = 0.f;
#pragma unroll
      for (int w2 = 0; w2 < 16; w2 += 4) {
        s0 += hpart[(w2 + 0) * H_DIM + tid];
        s1 += hpart[(w2 + 1) * H_DIM + tid];
        s2 += hpart[(w2 + 2) * H_DIM + tid];
        s3 += hpart[(w2 + 3) * H_DIM + tid];
      }
      float h = b1j + ((s0 + s1) + (s2 + s3));
      rhbuf[tid] = fmaxf(h, 0.0f);                             // ReLU(+b1)
    }
    __syncthreads();                                           // B2

    // --- MLP layer 2: q_c = b2_c + sum_j rh_j * W2[j,c]
    float q = b2c;
#pragma unroll
    for (int j = 0; j < H_DIM; ++j) q = fmaf(rhbuf[j], w2col[j], q);

    // --- EMA update + local sum of squares
    float r = 0.0f;
#pragma unroll
    for (int m = 0; m < M_DIM; ++m) {
      ema[m] = (1.0f - MOM) * ema[m] + MOM * (q * kf[m]);
      r = fmaf(ema[m], ema[m], r);
    }
    pw *= (1.0f - MOM);
    const float ib = 1.0f - pw;          // bias-correction 1 - (1-mom)^t

    // --- norm: reduce r over 512 threads
    r += __shfl_xor(r, 16, 32);
    r += __shfl_xor(r, 8, 32);
    r += __shfl_xor(r, 4, 32);
    r += __shfl_xor(r, 2, 32);
    r += __shfl_xor(r, 1, 32);
    if (lane == 0) wsum[w] = r;
    __syncthreads();                                           // B3
    float t0s = 0.f, t1s = 0.f, t2s = 0.f, t3s = 0.f;
#pragma unroll
    for (int w2 = 0; w2 < 16; w2 += 4) {
      t0s += wsum[w2 + 0];
      t1s += wsum[w2 + 1];
      t2s += wsum[w2 + 2];
      t3s += wsum[w2 + 3];
    }
    const float S = (t0s + t1s) + (t2s + t3s);

    // att_hat = ema*ib; denom = ib*sqrt(S)+1e-12; y = (att_hat/denom) . v
    const float f = ib / (ib * sqrtf(S) + 1e-12f);
    float ya = 0.0f;
#pragma unroll
    for (int m = 0; m < M_DIM; ++m) ya = fmaf(ema[m], vf[m], ya);
    const float ynew = f * ya;
    yprev = ynew;
    outstage[c * (TB + 1) + (t & (TB - 1))] = ynew;

    // --- coalesced flush of (C x TB) output tile every TB steps
    if ((t & (TB - 1)) == (TB - 1)) {
      __syncthreads();
      const int i = tid & (TB - 1);
      const int rbase = tid >> 4;          // 0..31
      const int tbase = t - (TB - 1);
#pragma unroll
      for (int ii = 0; ii < C_DIM / 32; ++ii) {
        int cr = rbase + 32 * ii;
        out[(size_t)cr * T_DIM + tbase + i] = outstage[cr * (TB + 1) + i];
      }
      __syncthreads();
    }
  }
}

// ---------------- host entry ----------------
extern "C" void kernel_launch(void* const* d_in, const int* in_sizes, int n_in,
                              void* d_out, int out_size, void* d_ws, size_t ws_size,
                              hipStream_t stream) {
  (void)in_sizes; (void)n_in; (void)out_size; (void)ws_size;
  const float* y  = (const float*)d_in[0];
  const float* K  = (const float*)d_in[1];
  const float* V  = (const float*)d_in[2];
  const float* W1 = (const float*)d_in[3];
  const float* b1 = (const float*)d_in[4];
  const float* W2 = (const float*)d_in[5];
  const float* b2 = (const float*)d_in[6];
  float* out = (float*)d_out;
  float* yT  = (float*)d_ws;   // needs T_DIM*C_DIM*4 = 16 MiB of scratch

  transpose_y_kernel<<<dim3(T_DIM / 32, C_DIM / 32), dim3(32, 8), 0, stream>>>(y, yT);
  ca_scan_kernel<<<dim3(1), dim3(512), 0, stream>>>(yT, K, V, W1, b1, W2, b2, out);
}
